// CMHSA_56487409877161
// MI455X (gfx1250) — matrix operand order repaired and checked by reading, not compiled, not run on hardware
//
#include <hip/hip_runtime.h>

typedef __attribute__((ext_vector_type(16))) __bf16 v16bf;
typedef __attribute__((ext_vector_type(8)))  __bf16 v8bf;
typedef __attribute__((ext_vector_type(8)))  float  v8f;
typedef __attribute__((ext_vector_type(4)))  unsigned int v4u;
typedef __attribute__((ext_vector_type(4)))  int v4i;
typedef __attribute__((ext_vector_type(8)))  int v8i;

#define B_     32
#define C_     512
#define HEADS_ 8
#define DH_    64
#define N_     1024
#define NBH_   (B_*HEADS_)
#define EPAD   1028   // LDS row stride (floats); 8*1028 % 64 != 0 -> no half-wave bank clash

// ---------------- WMMA helpers ----------------

__device__ inline v8f wmma_bf16(v16bf a, v16bf b, v8f c) {
  // D = A(16x32) * B(32x16) + C, f32 accumulate
  return __builtin_amdgcn_wmma_f32_16x16x32_bf16(false, a, false, b, (short)0, c, false, false);
}

// A fragment (16x32 bf16): lane m = lane%16 holds row m.
// lanes 0-15:  K = 0..15 (a[0..15])
// lanes 16-31: K = 16..31
__device__ inline v16bf make_a_frag(const __bf16* rowbase, int koff, int lhalf) {
  const v8bf lo = *reinterpret_cast<const v8bf*>(rowbase + koff + lhalf*16);
  const v8bf hi = *reinterpret_cast<const v8bf*>(rowbase + koff + lhalf*16 + 8);
  v16bf a;
#pragma unroll
  for (int i = 0; i < 8; ++i) { a[i] = lo[i]; a[8+i] = hi[i]; }
  return a;
}

// ---------------- small prep kernels ----------------

__global__ __launch_bounds__(256) void k_convert_w(const float* __restrict__ w,
                                                   __bf16* __restrict__ wb) {
  int t = blockIdx.x*256 + threadIdx.x;
  if (t < C_*C_) wb[t] = (__bf16)w[t];
}

// xT[b][n][c] = bf16(x[b][c][n])  -- LDS tiled transpose, coalesced both ways
__global__ __launch_bounds__(256) void k_transpose_x(const float* __restrict__ x,
                                                     __bf16* __restrict__ xt) {
  __shared__ float tile[32][33];
  int bx = blockIdx.x;                         // b*(16*32) + cblk*32 + nblk
  int nblk = bx & 31, cblk = (bx >> 5) & 15, b = bx >> 9;
  int col = threadIdx.x & 31, row4 = threadIdx.x >> 5;
  const float* src = x + ((size_t)b*C_ + cblk*32)*N_ + nblk*32;
#pragma unroll
  for (int r = 0; r < 32; r += 8)
    tile[r + row4][col] = src[(size_t)(r + row4)*N_ + col];
  __syncthreads();
  __bf16* dst = xt + ((size_t)b*N_ + nblk*32)*C_ + cblk*32;
#pragma unroll
  for (int r = 0; r < 32; r += 8)
    dst[(size_t)(r + row4)*C_ + col] = (__bf16)tile[col][r + row4];
}

// ---------------- startconv: qb[b][o][n] = bf16( W @ x + bias ) ----------------
__global__ __launch_bounds__(256) void k_startconv(const __bf16* __restrict__ wb,
                                                   const __bf16* __restrict__ xt,
                                                   const float*  __restrict__ bst,
                                                   __bf16* __restrict__ qb) {
  int bx   = blockIdx.x;                // 2048 = 32b * 8oblk * 8nblk
  int nblk = bx & 7, oblk = (bx >> 3) & 7, b = bx >> 6;
  int tid  = threadIdx.x, wave = tid >> 5, lane = tid & 31;
  int l16  = lane & 15, lhalf = lane >> 4;
  int o0 = oblk*64, n0 = nblk*128;
#pragma unroll
  for (int t = 0; t < 4; ++t) {
    int tile = wave + t*8;
    int ot = tile & 3, nt = tile >> 2;
    const __bf16* arow = wb + (size_t)(o0 + ot*16 + l16)*C_;
    const __bf16* brow = xt + ((size_t)b*N_ + n0 + nt*16 + l16)*C_;
    v8f acc = {0.f,0.f,0.f,0.f,0.f,0.f,0.f,0.f};
#pragma unroll
    for (int k = 0; k < 16; ++k) {
      v16bf a = make_a_frag(arow, k*32, lhalf);
      v16bf bb = *reinterpret_cast<const v16bf*>(brow + k*32 + lhalf*16);
      acc = wmma_bf16(a, bb, acc);
    }
    int ncol = n0 + nt*16 + l16;
#pragma unroll
    for (int j = 0; j < 8; ++j) {
      int orow = o0 + ot*16 + j + lhalf*8;
      qb[((size_t)b*C_ + orow)*N_ + ncol] = (__bf16)(acc[j] + bst[orow]);
    }
  }
}

// ---------------- pack (tiled transpose): qt[bh][n][d], qpt = q/sqrt(128)+pos ----------------
__global__ __launch_bounds__(256) void k_pack(const __bf16* __restrict__ qb,
                                              const float* __restrict__ relh,
                                              const float* __restrict__ relw,
                                              __bf16* __restrict__ qt,
                                              __bf16* __restrict__ qpt) {
  __shared__ float tile[32][33];
  int bx = blockIdx.x;                         // bh*64 + dblk*32 + nblk
  int nblk = bx & 31, dblk = (bx >> 5) & 1, bh = bx >> 6;
  int h = bh & 7;
  int col = threadIdx.x & 31, row4 = threadIdx.x >> 5;
  const __bf16* src = qb + ((size_t)bh*64 + dblk*32)*N_ + nblk*32;
#pragma unroll
  for (int r = 0; r < 32; r += 8)
    tile[r + row4][col] = (float)src[(size_t)(r + row4)*N_ + col];
  __syncthreads();
  int d = dblk*32 + col;
#pragma unroll
  for (int r = 0; r < 32; r += 8) {
    int nr = r + row4;
    int n = nblk*32 + nr;
    float qv = tile[col][nr];
    size_t oidx = ((size_t)bh*N_ + n)*64 + d;
    qt[oidx] = (__bf16)qv;
    int wcol = n >> 5, hrow = n & 31;
    float p = relh[(h*64 + d)*32 + hrow] + relw[(h*64 + d)*32 + wcol];
    qpt[oidx] = (__bf16)(qv*0.08838834764831845f + p);   // 1/sqrt(128)
  }
}

// qsum[bh][d] = sum_n q[d][n] -- coalesced reads from transposed layout
__global__ __launch_bounds__(64) void k_qsum(const __bf16* __restrict__ qt,
                                             float* __restrict__ qsum) {
  int bh = blockIdx.x, d = threadIdx.x;
  const __bf16* p = qt + (size_t)bh*N_*64 + d;
  float s = 0.f;
  for (int n = 0; n < N_; ++n) s += (float)p[(size_t)n*64];
  qsum[bh*64 + d] = s;
}

// ---------------- attention: E = Q'^T K ; softmax ; U = P Q^T ----------------
__global__ __launch_bounds__(256) void k_attn(const __bf16* __restrict__ qt,
                                              const __bf16* __restrict__ qpt,
                                              const __bf16* __restrict__ qb,
                                              float* __restrict__ U,
                                              float* __restrict__ ssq) {
  __shared__ float E[32*EPAD];
  __shared__ float red[32];
  __shared__ __bf16 QS[32*64];          // staged Q' block (TDM target)
  int blk  = blockIdx.x;                // 8192 = 256 bh * 32 qblk
  int qblk = blk & 31, bh = blk >> 5;
  int q0   = qblk*32;
  int tid  = threadIdx.x, wave = tid >> 5, lane = tid & 31;
  int l16  = lane & 15, lhalf = lane >> 4;
  const __bf16* qtB  = qt  + (size_t)bh*N_*64;
  const __bf16* qptB = qpt + (size_t)bh*N_*64;

  // Stage Q' (32 rows x 64 d, bf16) into LDS via the Tensor Data Mover.
#if defined(__gfx1250__) && __has_builtin(__builtin_amdgcn_tensor_load_to_lds)
  if (wave == 0) {
    unsigned lds_off = (unsigned)(size_t)(void*)QS;       // low 32 bits = LDS offset
    unsigned long long ga = (unsigned long long)(size_t)(qptB + (size_t)q0*64);
    v4u g0;
    g0[0] = 1u;                                           // count=1, no gather
    g0[1] = lds_off;                                      // lds_addr
    g0[2] = (unsigned)(ga & 0xFFFFFFFFu);                 // global_addr[31:0]
    g0[3] = (unsigned)((ga >> 32) & 0x01FFFFFFu) | (2u << 30);  // addr[56:32] | type=2
    v8i g1;
    g1[0] = (int)(1u << 16);        // workgroup_mask=0, data_size=1 (2 bytes)
    g1[1] = (int)(64u << 16);       // tensor_dim0 = 64   (bits 79:48, low 16)
    g1[2] = (int)(1024u << 16);     // tensor_dim1 = 1024 (bits 111:80, low 16)
    g1[3] = (int)(64u << 16);       // tile_dim0 = 64     (bits 127:112)
    g1[4] = 32;                     // tile_dim1 = 32, tile_dim2 = 0
    g1[5] = 64;                     // tensor_dim0_stride = 64 (low 32)
    g1[6] = 0;
    g1[7] = 0;
    v4i gz = {0,0,0,0};
#if __clang_major__ >= 23
    v8i gz8 = {0,0,0,0,0,0,0,0};
    __builtin_amdgcn_tensor_load_to_lds(g0, g1, gz, gz, gz8, 0);
#else
    __builtin_amdgcn_tensor_load_to_lds(g0, g1, gz, gz, 0);
#endif
    __builtin_amdgcn_s_wait_tensorcnt(0);
  }
#else
  for (int i = tid; i < 32*64; i += 256) QS[i] = qptB[(size_t)q0*64 + i];
#endif
  __syncthreads();

  // Phase A: energy. Each wave owns unique k-tiles, computes BOTH query tiles per B load.
  {
    v16bf A00 = make_a_frag(QS + (size_t)(l16)*64,      0,  lhalf);
    v16bf A01 = make_a_frag(QS + (size_t)(l16)*64,      32, lhalf);
    v16bf A10 = make_a_frag(QS + (size_t)(16 + l16)*64, 0,  lhalf);
    v16bf A11 = make_a_frag(QS + (size_t)(16 + l16)*64, 32, lhalf);
    for (int kt = wave; kt < 64; kt += 8) {
      const __bf16* brow = qtB + (size_t)(kt*16 + l16)*64;
      v16bf B0 = *reinterpret_cast<const v16bf*>(brow + lhalf*16);
      v16bf B1 = *reinterpret_cast<const v16bf*>(brow + 32 + lhalf*16);
      v8f acc0 = {0.f,0.f,0.f,0.f,0.f,0.f,0.f,0.f};
      v8f acc1 = {0.f,0.f,0.f,0.f,0.f,0.f,0.f,0.f};
      acc0 = wmma_bf16(A00, B0, acc0);
      acc0 = wmma_bf16(A01, B1, acc0);
      acc1 = wmma_bf16(A10, B0, acc1);
      acc1 = wmma_bf16(A11, B1, acc1);
#pragma unroll
      for (int j = 0; j < 8; ++j) {
        E[(j + lhalf*8)*EPAD + kt*16 + l16]        = acc0[j];
        E[(16 + j + lhalf*8)*EPAD + kt*16 + l16]   = acc1[j];
      }
    }
  }
  __syncthreads();

  // Phase B: row softmax + sum of squares (8 lanes per row, stride-8 columns)
  {
    int row = tid >> 3, seg = tid & 7;
    float* er = E + row*EPAD;
    float mx = -3.4e38f;
    for (int i = 0; i < 128; ++i) mx = fmaxf(mx, er[seg + 8*i]);
#pragma unroll
    for (int m = 1; m < 8; m <<= 1) mx = fmaxf(mx, __shfl_xor(mx, m, 32));
    float s = 0.f;
    for (int i = 0; i < 128; ++i) {
      float p = __expf(er[seg + 8*i] - mx);
      er[seg + 8*i] = p; s += p;
    }
#pragma unroll
    for (int m = 1; m < 8; m <<= 1) s += __shfl_xor(s, m, 32);
    float inv = 1.f / s;
    float sq = 0.f;
    for (int i = 0; i < 128; ++i) {
      float p = er[seg + 8*i] * inv;
      er[seg + 8*i] = p; sq += p*p;
    }
#pragma unroll
    for (int m = 1; m < 8; m <<= 1) sq += __shfl_xor(sq, m, 32);
    if (seg == 0) red[row] = sq;
  }
  __syncthreads();
  if (tid == 0) {
    float t = 0.f;
    for (int r = 0; r < 32; ++r) t += red[r];
    ssq[bh*32 + qblk] = t;
  }

  // Phase C: U[m][d] = sum_n P[m][n] * q[d][n].  8 waves = 2 mtiles x 4 dtiles.
  {
    int mt = wave & 1, dt = wave >> 1;
    v8f acc = {0.f,0.f,0.f,0.f,0.f,0.f,0.f,0.f};
    const float*  prow = E  + (size_t)(mt*16 + l16)*EPAD;
    const __bf16* brow = qb + ((size_t)bh*64 + dt*16 + l16)*N_;
    for (int kc = 0; kc < 32; ++kc) {
      const float* ep = prow + kc*32 + lhalf*16;
      v16bf a;
#pragma unroll
      for (int i = 0; i < 8; ++i) { a[i] = (__bf16)ep[i]; a[8+i] = (__bf16)ep[8+i]; }
      v16bf bb = *reinterpret_cast<const v16bf*>(brow + kc*32 + lhalf*16);
      acc = wmma_bf16(a, bb, acc);
    }
    float* ub = U + ((size_t)bh*N_ + q0 + mt*16)*64 + dt*16;
#pragma unroll
    for (int j = 0; j < 8; ++j)
      ub[(size_t)(j + lhalf*8)*64 + l16] = acc[j];
  }
}

// ---------------- reductions: rs[bh], t[bh][e] = sum_d Wl[e][d]*qsum[d] ----------------
__global__ __launch_bounds__(64) void k_reduce(const float* __restrict__ ssq,
                                               const float* __restrict__ qsum,
                                               const float* __restrict__ wl,
                                               float* __restrict__ tbuf,
                                               float* __restrict__ rsbuf) {
  int bh = blockIdx.x, e = threadIdx.x;
  const float* qs = qsum + bh*64;
  float t = 0.f;
  for (int d = 0; d < 64; ++d) t += wl[e*64 + d] * qs[d];
  tbuf[bh*64 + e] = t;
  if (e == 0) {
    float s = 0.f;
    for (int i = 0; i < 32; ++i) s += ssq[bh*32 + i];
    const float mu = 1.f/1024.f;
    float var = s * (1.f/1048576.f) - mu*mu;
    rsbuf[bh] = rsqrtf(var + 1e-5f);
  }
}

// ---------------- final: out = rs*(U @ Wl^T) - rs*mu*t + b_lin + x ----------------
__global__ __launch_bounds__(256) void k_final(const float* __restrict__ U,
                                               const float* __restrict__ wl,
                                               const float* __restrict__ bl,
                                               const float* __restrict__ tbuf,
                                               const float* __restrict__ rsbuf,
                                               const float* __restrict__ x,
                                               float* __restrict__ out) {
  int bx = blockIdx.x;                  // 2048 = 256 bh * 8 mblk
  int mblk = bx & 7, bh = bx >> 3;
  int b = bh >> 3, h = bh & 7;
  int tid = threadIdx.x, wave = tid >> 5, lane = tid & 31;
  int l16 = lane & 15, lhalf = lane >> 4;
  int m0 = mblk*128 + wave*16;
  float rs = rsbuf[bh];
  const float mu = 1.f/1024.f;
  const float* urow = U + ((size_t)bh*N_ + m0 + l16)*64;
#pragma unroll
  for (int et = 0; et < 4; ++et) {
    v8f acc = {0.f,0.f,0.f,0.f,0.f,0.f,0.f,0.f};
#pragma unroll
    for (int c = 0; c < 2; ++c) {
      v16bf a;
      const float* ap = urow + c*32 + lhalf*16;
#pragma unroll
      for (int i = 0; i < 8; ++i) { a[i] = (__bf16)ap[i]; a[8+i] = (__bf16)ap[8+i]; }
      v16bf bb;
      const float* bp = wl + (et*16 + l16)*64 + c*32 + lhalf*16;
#pragma unroll
      for (int i = 0; i < 16; ++i) bb[i] = (__bf16)bp[i];
      acc = wmma_bf16(a, bb, acc);
    }
    int e = et*16 + l16;
    int cch = h*64 + e;
    float corr = -rs*mu*tbuf[bh*64 + e] + bl[e];
    const float* xr  = x   + ((size_t)b*C_ + cch)*N_;
    float*       orw = out + ((size_t)b*C_ + cch)*N_;
#pragma unroll
    for (int j = 0; j < 8; ++j) {
      int m = m0 + j + lhalf*8;
      orw[m] = rs*acc[j] + corr + xr[m];
    }
  }
}

// ---------------- launch ----------------

extern "C" void kernel_launch(void* const* d_in, const int* in_sizes, int n_in,
                              void* d_out, int out_size, void* d_ws, size_t ws_size,
                              hipStream_t stream) {
  (void)in_sizes; (void)n_in; (void)out_size; (void)ws_size;
  const float* x   = (const float*)d_in[0];
  const float* wst = (const float*)d_in[1];
  const float* bst = (const float*)d_in[2];
  const float* rlh = (const float*)d_in[3];
  const float* rlw = (const float*)d_in[4];
  const float* wl  = (const float*)d_in[5];
  const float* bl  = (const float*)d_in[6];
  float* out = (float*)d_out;

  char* ws = (char*)d_ws;
  size_t off = 0;
  auto take = [&](size_t bytes) -> char* {
    char* p = ws + off;
    off = (off + bytes + 255) & ~(size_t)255;
    return p;
  };
  __bf16* wbf  = (__bf16*)take((size_t)C_*C_*2);            // 0.5 MB
  __bf16* xt   = (__bf16*)take((size_t)B_*N_*C_*2);         // 32 MB
  __bf16* qb   = (__bf16*)take((size_t)B_*C_*N_*2);         // 32 MB
  __bf16* qt   = (__bf16*)take((size_t)NBH_*N_*64*2);       // 32 MB
  __bf16* qpt  = (__bf16*)take((size_t)NBH_*N_*64*2);       // 32 MB
  float*  U    = (float*)take((size_t)NBH_*N_*64*4);        // 64 MB
  float*  qsum = (float*)take((size_t)NBH_*64*4);
  float*  ssq  = (float*)take((size_t)NBH_*32*4);
  float*  rsb  = (float*)take((size_t)NBH_*4);
  float*  tb   = (float*)take((size_t)NBH_*64*4);

  k_convert_w  <<<1024,  256, 0, stream>>>(wst, wbf);
  k_transpose_x<<<16384, 256, 0, stream>>>(x, xt);
  k_startconv  <<<2048,  256, 0, stream>>>(wbf, xt, bst, qb);
  k_pack       <<<16384, 256, 0, stream>>>(qb, rlh, rlw, qt, qpt);
  k_qsum       <<<NBH_,  64,  0, stream>>>(qt, qsum);
  k_attn       <<<NBH_*32, 256, 0, stream>>>(qt, qpt, qb, U, ssq);
  k_reduce     <<<NBH_,  64,  0, stream>>>(ssq, qsum, wl, tb, rsb);
  k_final      <<<NBH_*8, 256, 0, stream>>>(U, wl, bl, tb, rsb, x, out);
}
